// Dense_test_1layer_30940944400811
// MI455X (gfx1250) — compile-verified, matrix operand order checked by reading, not run on hardware
//
#include <hip/hip_runtime.h>
#include <hip/hip_bf16.h>

typedef __attribute__((ext_vector_type(16))) _Float16 v16h;
typedef __attribute__((ext_vector_type(4)))  _Float16 v4h;
typedef __attribute__((ext_vector_type(8)))  float    v8f;

#define B_    128
#define T_    500
#define DIN   700
#define N_    64
#define C_    20
#define INSZ  764   // DIN + N
#define KPAD  768   // padded K (multiple of 32)
#define NKT   24    // KPAD/32 k-steps
#define NBCOL 256   // N * BR dendritic columns
#define NNT   16    // NBCOL/16 n-tiles
#define XSTRIDE 704 // staging row stride in floats (176 chunks of 4)

// --- WMMA 16-bit fragment address helpers (ISA 7.12.2 16-bit A 16x32 layout) ---
__device__ __forceinline__ void kSplit(int kk, int& laneHi, int& e) {
    laneHi = (kk >> 3) & 1;                 // K 8..15 / 24..31 live in lanes 16..31
    e = (kk & 7) + ((kk & 16) ? 8 : 0);     // element slot within the lane's 16 halfs
}
__device__ __forceinline__ int fragAddr(int row16, int laneHiK, int e) {
    return (row16 + (laneHiK << 4)) * 16 + e;   // [lane][16] within a 512-half frag
}

// --- CDNA5 async global->LDS copy (ASYNCcnt path, ISA 10 / 15.18.3 op 98) ---
__device__ __forceinline__ void async_b128_to_lds(const float* g, const float* l) {
    unsigned int lds = (unsigned int)(size_t)l;   // low 32 bits of generic = LDS offset
    asm volatile("global_load_async_to_lds_b128 %0, %1, off"
                 :: "v"(lds), "v"(g) : "memory");
}
__device__ __forceinline__ void wait_async_le(void) {   // allow 11 newest in flight
    asm volatile("s_wait_asynccnt 11" ::: "memory");
}
__device__ __forceinline__ void wait_async_0(void) {
    asm volatile("s_wait_asynccnt 0" ::: "memory");
}

// ---------------- pre-pack W1*mask (f32) -> f16 B fragments in workspace ----------------
__global__ __launch_bounds__(256)
void pack_w1(const float* __restrict__ W1, const float* __restrict__ mask,
             _Float16* __restrict__ Bbuf) {
    int id = blockIdx.x * 256 + threadIdx.x;   // over 256*768 elements
    int nb = id / KPAD;                        // dendritic column 0..255
    int k  = id % KPAD;                        // padded K
    float v = 0.f;
    if (k < INSZ) v = W1[nb * INSZ + k] * mask[nb * INSZ + k];
    int nt = nb >> 4, ln = nb & 15;
    int kt = k >> 5,  kk = k & 31;
    int lh, e; kSplit(kk, lh, e);
    Bbuf[(size_t)(kt * NNT + nt) * 512 + (size_t)fragAddr(ln, lh, e)] = (_Float16)v;
}

// ---------------- main recurrent kernel: one workgroup = 16 batch rows ----------------
__global__ __launch_bounds__(256)
void dhsnn_kernel(const float* __restrict__ x,      const float* __restrict__ b1,
                  const float* __restrict__ tau_m1, const float* __restrict__ tau_n1,
                  const float* __restrict__ W2,     const float* __restrict__ b2,
                  const float* __restrict__ tau_m2, const float* __restrict__ mem1_0,
                  const float* __restrict__ mem2_0, const _Float16* __restrict__ Bbuf,
                  float* __restrict__ out) {
    __shared__ _Float16 Abuf[NKT * 512];        // 24 KB: A operand in fragment layout
    __shared__ float xstage[2][16 * XSTRIDE];   // 2 x 44 KB: async x_t staging
    __shared__ float W2s[C_ * N_];              // readout weights
    __shared__ float b2s[C_], a2s[C_];          // readout bias, sigmoid(tau_m2)
    __shared__ float mem2s[16 * C_];            // readout membrane
    __shared__ float spkS[16 * N_];             // spikes as f32 for readout GEMV
    __shared__ float outacc[16 * C_];           // softmax accumulator

    const int tid  = threadIdx.x;
    const int lane = tid & 31;
    const int wave = tid >> 5;
    const int r0   = blockIdx.x * 16;           // batch rows [r0, r0+16)
    const int laneHiM = (lane >= 16) ? 8 : 0;   // this lane's C-frag rows are i+laneHiM
    const int xrow = tid >> 4;                  // staging row owned by this thread
    const int xcol = tid & 15;                  // chunk phase owned by this thread

    for (int i = tid; i < NKT * 512; i += 256) Abuf[i] = (_Float16)0.f;  // K-pad + spk=0
    for (int i = tid; i < 16 * C_;  i += 256) {
        outacc[i] = 0.f;
        mem2s[i]  = mem2_0[(r0 + i / C_) * C_ + (i % C_)];
    }
    for (int i = tid; i < C_ * N_;  i += 256) W2s[i] = W2[i];
    for (int i = tid; i < 16 * N_;  i += 256) spkS[i] = 0.f;
    if (tid < C_) { b2s[tid] = b2[tid]; a2s[tid] = 1.f / (1.f + __expf(-tau_m2[tid])); }

    // per-lane per-tile constants + register-resident state
    const int nt0 = wave * 2;                   // this wave owns n-tiles nt0, nt0+1
    float beta[2], coefB[2], cb1[2], alpha[2];
    float mem1[2][8], spkp[2][8];
    v8f dinp[2];
#pragma unroll
    for (int j = 0; j < 2; ++j) {
        int c = (nt0 + j) * 16 + (lane & 15);   // flat (neuron, branch) column
        int n = c >> 2;
        beta[j]  = 1.f / (1.f + __expf(-tau_n1[c]));
        coefB[j] = 1.f - beta[j];
        cb1[j]   = coefB[j] * b1[c];
        alpha[j] = 1.f / (1.f + __expf(-tau_m1[n]));
#pragma unroll
        for (int i = 0; i < 8; ++i) {
            mem1[j][i] = mem1_0[(r0 + i + laneHiM) * N_ + n];
            spkp[j][i] = 0.f;
            dinp[j][i] = 0.f;
        }
    }

    // prologue: async-prefetch x_0 into staging buffer 0 (11 uniform ops/lane)
    {
        const float* gr = x + ((size_t)(r0 + xrow) * T_ + 0) * DIN;
        const float* lr = &xstage[0][xrow * XSTRIDE];
        for (int c = xcol; c < 176; c += 16) {
            int off = (c < 175) ? c * 4 : 0;            // clamp dummy tail chunk
            int lof = (c < 175) ? c * 4 : 700;
            async_b128_to_lds(gr + off, lr + lof);
        }
    }
    __syncthreads();

    for (int t = 0; t < T_; ++t) {
        const int buf  = t & 1;
        const int nbuf = buf ^ 1;

        // ---- issue async prefetch of x_{t+1} into the other staging buffer ----
        if (t + 1 < T_) {
            const float* gr = x + ((size_t)(r0 + xrow) * T_ + (t + 1)) * DIN;
            const float* lr = &xstage[nbuf][xrow * XSTRIDE];
            for (int c = xcol; c < 176; c += 16) {
                int off = (c < 175) ? c * 4 : 0;
                int lof = (c < 175) ? c * 4 : 700;
                async_b128_to_lds(gr + off, lr + lof);
            }
            wait_async_le();    // drain step-t copies; keep the 11 new ones in flight
        } else {
            wait_async_0();
        }

        // ---- convert staged x_t (f32, LDS) into f16 A fragments ----
        {
            const float* xr = &xstage[buf][xrow * XSTRIDE];
            for (int c4 = xcol; c4 < DIN / 4; c4 += 16) {
                int d0 = c4 * 4;
                float4 v = *(const float4*)(xr + d0);
                int kt = d0 >> 5, kk = d0 & 31;
                int lh, e; kSplit(kk, lh, e);           // 4-aligned run stays in one group
                v4h h; h.x = (_Float16)v.x; h.y = (_Float16)v.y;
                       h.z = (_Float16)v.z; h.w = (_Float16)v.w;
                *(v4h*)(&Abuf[kt * 512 + fragAddr(xrow, lh, e)]) = h;   // ds_store_b64
            }
        }
        __syncthreads();

        // ---- layer-1 GEMM: (16 x 768) @ (768 x 32-per-wave) via WMMA f16 ----
        v8f acc0 = {}, acc1 = {};
#pragma unroll 4
        for (int kt = 0; kt < NKT; ++kt) {
            v16h a  = *(const v16h*)(&Abuf[kt * 512 + lane * 16]);
            v16h bA = *(const v16h*)(Bbuf + (size_t)(kt * NNT + nt0    ) * 512 + lane * 16);
            v16h bB = *(const v16h*)(Bbuf + (size_t)(kt * NNT + nt0 + 1) * 512 + lane * 16);
            acc0 = __builtin_amdgcn_wmma_f32_16x16x32_f16(false, a, false, bA,
                                                          (short)0, acc0, false, false);
            acc1 = __builtin_amdgcn_wmma_f32_16x16x32_f16(false, a, false, bB,
                                                          (short)0, acc1, false, false);
        }

        // ---- dendrite filter + branch sum + soma LIF (all in registers) ----
        float spk[2][8];
#pragma unroll
        for (int j = 0; j < 2; ++j) {
#pragma unroll
            for (int i = 0; i < 8; ++i) {
                float a = (j == 0) ? acc0[i] : acc1[i];
                float d = beta[j] * dinp[j][i] + coefB[j] * a + cb1[j];
                dinp[j][i] = d;
                // sum 4 branches = 4 adjacent lanes (column groups of 4)
                d += __shfl_xor(d, 1, 32);
                d += __shfl_xor(d, 2, 32);
                float m = alpha[j] * mem1[j][i] + (1.f - alpha[j]) * d - spkp[j][i];
                mem1[j][i] = m;
                float s = (m > 1.f) ? 1.f : 0.f;
                spk[j][i]  = s;
                spkp[j][i] = s;
            }
        }
        __syncthreads();   // all A-fragment reads complete before spike republish

        // ---- publish spikes: into A fragments (cols 700..763) and spkS (f32) ----
        if ((lane & 3) == 0) {
#pragma unroll
            for (int j = 0; j < 2; ++j) {
                int n  = ((nt0 + j) * 16 + (lane & 15)) >> 2;   // neuron owned by group
                int cc = DIN + n;
                int kt = cc >> 5, kk = cc & 31;
                int lh, e; kSplit(kk, lh, e);
#pragma unroll
                for (int i = 0; i < 8; ++i) {
                    int m = i + laneHiM;
                    Abuf[kt * 512 + fragAddr(m, lh, e)] = (_Float16)spk[j][i];
                    spkS[m * N_ + n] = spk[j][i];
                }
            }
        }
        __syncthreads();

        // ---- readout: mem2 = a2*mem2 + (1-a2)*(spk @ W2^T + b2) ----
        for (int p = tid; p < 16 * C_; p += 256) {
            int row = p / C_, cls = p % C_;
            float dot = 0.f;
            const float* sr = &spkS[row * N_];
            const float* wr = &W2s[cls * N_];
#pragma unroll 8
            for (int n = 0; n < N_; ++n) dot += sr[n] * wr[n];
            float a2 = a2s[cls];
            mem2s[p] = a2 * mem2s[p] + (1.f - a2) * (dot + b2s[cls]);
        }
        __syncthreads();

        // ---- softmax over classes, accumulate for t > 10 ----
        if (tid < 16 && t > 10) {
            float mx = -1e30f;
            for (int c = 0; c < C_; ++c) mx = fmaxf(mx, mem2s[tid * C_ + c]);
            float sum = 0.f, ex[C_];
            for (int c = 0; c < C_; ++c) { ex[c] = __expf(mem2s[tid * C_ + c] - mx); sum += ex[c]; }
            float inv = 1.f / sum;
            for (int c = 0; c < C_; ++c) outacc[tid * C_ + c] += ex[c] * inv;
        }
        __syncthreads();
    }

    for (int p = tid; p < 16 * C_; p += 256) out[r0 * C_ + p] = outacc[p];
}

extern "C" void kernel_launch(void* const* d_in, const int* in_sizes, int n_in,
                              void* d_out, int out_size, void* d_ws, size_t ws_size,
                              hipStream_t stream) {
    const float* x      = (const float*)d_in[0];
    const float* W1     = (const float*)d_in[1];
    const float* b1     = (const float*)d_in[2];
    const float* mask   = (const float*)d_in[3];
    const float* tau_m1 = (const float*)d_in[4];
    const float* tau_n1 = (const float*)d_in[5];
    const float* W2     = (const float*)d_in[6];
    const float* b2     = (const float*)d_in[7];
    const float* tau_m2 = (const float*)d_in[8];
    const float* mem1_0 = (const float*)d_in[9];
    const float* mem2_0 = (const float*)d_in[10];
    float* out = (float*)d_out;

    _Float16* Bbuf = (_Float16*)d_ws;   // 768*256*2 = 384 KB of workspace

    pack_w1<<<(KPAD * NBCOL) / 256, 256, 0, stream>>>(W1, mask, Bbuf);
    dhsnn_kernel<<<B_ / 16, 256, 0, stream>>>(x, b1, tau_m1, tau_n1, W2, b2, tau_m2,
                                              mem1_0, mem2_0, Bbuf, out);
}
